// OnLSTMCell_79267916415623
// MI455X (gfx1250) — matrix operand order repaired and checked by reading, not compiled, not run on hardware
//
#include <hip/hip_runtime.h>
#include <hip/hip_bf16.h>
#include <cstdint>
#include <cstddef>

// ---------------------------------------------------------------------------
// ON-LSTM cell for MI455X (gfx1250, wave32, WMMA).
// GEMM in bf16x3 split precision (hi/lo) via v_wmma_f32_16x16x32_bf16.
// Tiles staged with CDNA5 async global->LDS (GLOBAL_LOAD_ASYNC_TO_LDS_B128),
// double-buffered in LDS so copy latency overlaps the WMMA stream.
// ---------------------------------------------------------------------------

typedef __attribute__((ext_vector_type(16))) __bf16 v16bf;
typedef __attribute__((ext_vector_type(8)))  __bf16 v8bf;
typedef __attribute__((ext_vector_type(8)))  float  v8f;

#define BSZ     8192
#define IN_DIM  1024
#define HID     1024
#define KDIM    2048        // IN + HID
#define NLSTM   4096        // 4*HID
#define NLVL    128         // 2*LVL
#define NTOT    4224        // NLSTM + NLVL
#define LVL     64

#define LDS_ROW     40                  // 32 halves + 8 pad (conflict-free b128)
#define SMEM_HALVES (128 * LDS_ROW)     // halves per tile array
#define BUF_HALVES  (4 * SMEM_HALVES)   // Ah|Al|Bh|Bl per buffer
#define BUF_BYTES   (2u * BUF_HALVES)

// ---- bf16 split helpers ----------------------------------------------------
__device__ __forceinline__ unsigned short f2bf(float f) {
  unsigned int u = __float_as_uint(f);
  u += 0x7FFFu + ((u >> 16) & 1u);           // round-to-nearest-even
  return (unsigned short)(u >> 16);
}
__device__ __forceinline__ float bf2f(unsigned short h) {
  return __uint_as_float(((unsigned int)h) << 16);
}
__device__ __forceinline__ void split2(float x, unsigned short& hi, unsigned short& lo) {
  hi = f2bf(x);
  lo = f2bf(x - bf2f(hi));
}

// ---- pack A = concat(input, h_prev) into bf16 hi/lo planes -----------------
__global__ __launch_bounds__(256) void pack_A(const float* __restrict__ input,
                                              const float* __restrict__ h_prev,
                                              unsigned short* __restrict__ hi,
                                              unsigned short* __restrict__ lo) {
  size_t i = ((size_t)blockIdx.x * 256 + threadIdx.x) * 4;
  if (i >= (size_t)BSZ * KDIM) return;
  size_t row = i >> 11;                       // / KDIM
  int k = (int)(i & (KDIM - 1));
  const float* src = (k < IN_DIM) ? (input + row * IN_DIM + k)
                                  : (h_prev + row * HID + (k - IN_DIM));
  float4 v = *(const float4*)src;
  unsigned short h0, h1, h2, h3, l0, l1, l2, l3;
  split2(v.x, h0, l0); split2(v.y, h1, l1);
  split2(v.z, h2, l2); split2(v.w, h3, l3);
  uint2 ph, pl;
  ph.x = (unsigned)h0 | ((unsigned)h1 << 16);
  ph.y = (unsigned)h2 | ((unsigned)h3 << 16);
  pl.x = (unsigned)l0 | ((unsigned)l1 << 16);
  pl.y = (unsigned)l2 | ((unsigned)l3 << 16);
  *(uint2*)(hi + i) = ph;
  *(uint2*)(lo + i) = pl;
}

// ---- pack B = concat(W_lstm, W_level) rows, K-contiguous -------------------
__global__ __launch_bounds__(256) void pack_B(const float* __restrict__ W_lstm,
                                              const float* __restrict__ W_level,
                                              unsigned short* __restrict__ hi,
                                              unsigned short* __restrict__ lo) {
  size_t i = ((size_t)blockIdx.x * 256 + threadIdx.x) * 4;
  if (i >= (size_t)NTOT * KDIM) return;
  size_t row = i / KDIM;
  int k = (int)(i % KDIM);
  const float* src = (row < NLSTM) ? (W_lstm + row * KDIM + k)
                                   : (W_level + (row - NLSTM) * KDIM + k);
  float4 v = *(const float4*)src;
  unsigned short h0, h1, h2, h3, l0, l1, l2, l3;
  split2(v.x, h0, l0); split2(v.y, h1, l1);
  split2(v.z, h2, l2); split2(v.w, h3, l3);
  uint2 ph, pl;
  ph.x = (unsigned)h0 | ((unsigned)h1 << 16);
  ph.y = (unsigned)h2 | ((unsigned)h3 << 16);
  pl.x = (unsigned)l0 | ((unsigned)l1 << 16);
  pl.y = (unsigned)l2 | ((unsigned)l3 << 16);
  *(uint2*)(hi + i) = ph;
  *(uint2*)(lo + i) = pl;
}

// ---- async stage of one K-tile (8 x B128 per thread, no VGPR round-trip) ---
__device__ __forceinline__ void stage_async(
    unsigned ldsAh, unsigned ldsAl, unsigned ldsBh, unsigned ldsBl,
    unsigned aOff, unsigned bOff,
    const unsigned short* Ahi, const unsigned short* Alo,
    const unsigned short* Bhi, const unsigned short* Blo) {
  asm volatile(
      "global_load_async_to_lds_b128 %0, %4, %6\n\t"
      "global_load_async_to_lds_b128 %0, %4, %6 offset:16\n\t"
      "global_load_async_to_lds_b128 %1, %4, %7\n\t"
      "global_load_async_to_lds_b128 %1, %4, %7 offset:16\n\t"
      "global_load_async_to_lds_b128 %2, %5, %8\n\t"
      "global_load_async_to_lds_b128 %2, %5, %8 offset:16\n\t"
      "global_load_async_to_lds_b128 %3, %5, %9\n\t"
      "global_load_async_to_lds_b128 %3, %5, %9 offset:16"
      :
      : "v"(ldsAh), "v"(ldsAl), "v"(ldsBh), "v"(ldsBl),
        "v"(aOff), "v"(bOff),
        "s"(Ahi), "s"(Alo), "s"(Bhi), "s"(Blo)
      : "memory");
}

// ---- GEMM: C(8192x4224) = A(8192x2048) * B(4224x2048)^T, bf16x3 -----------
// Block tile 128x128, K-tile 32, double-buffered LDS (2 x 40KB).
// 8 waves: wave grid 4(M) x 2(N), each wave owns 2x4 16x16 sub-tiles,
// 24 v_wmma_f32_16x16x32_bf16 per K-step.
union BFV { v16bf v; v8bf h[2]; };

__global__ __launch_bounds__(256) void gemm_bf16x3(
    const unsigned short* __restrict__ Ahi, const unsigned short* __restrict__ Alo,
    const unsigned short* __restrict__ Bhi, const unsigned short* __restrict__ Blo,
    float* __restrict__ C) {
  __shared__ unsigned short smem[2 * BUF_HALVES];    // double buffer

  const int tid  = threadIdx.x;
  const int lane = tid & 31;
  const int wave = tid >> 5;
  const int wm   = wave >> 1;     // 0..3  -> M offset 32*wm
  const int wn   = wave & 1;      // 0..1  -> N offset 64*wn
  const int fr   = lane & 15;     // row/col within 16x16 fragment
  const int hk   = lane >> 4;     // half-wave selector (ISA A/B layouts)

  const int m0 = blockIdx.y * 128;
  const int n0 = blockIdx.x * 128;

  // staging map: 2 threads per row, 16 halves (32B) each, contiguous in both
  // global and LDS -> two B128 async copies at offset 0 / 16.
  const int srow  = tid >> 1;
  const int skoff = (tid & 1) << 4;

  // LDS byte addresses (generic pointer low 32 bits == LDS offset)
  const unsigned ldsOff  = 2u * (unsigned)(srow * LDS_ROW + skoff);
  const unsigned ldsBase = (unsigned)(uintptr_t)(&smem[0]);
  const unsigned ldsAh = ldsBase + ldsOff;
  const unsigned ldsAl = ldsBase + 2u * SMEM_HALVES + ldsOff;
  const unsigned ldsBh = ldsBase + 4u * SMEM_HALVES + ldsOff;
  const unsigned ldsBl = ldsBase + 6u * SMEM_HALVES + ldsOff;

  // global byte offsets (advance 64B per K-tile of 32 halves)
  unsigned aOff = ((unsigned)(m0 + srow) * KDIM + (unsigned)skoff) * 2u;
  unsigned bOff = ((unsigned)(n0 + srow) * KDIM + (unsigned)skoff) * 2u;

  v8f acc[2][4] = {};

  // prologue: stage tile 0 into buffer 0
  stage_async(ldsAh, ldsAl, ldsBh, ldsBl, aOff, bOff, Ahi, Alo, Bhi, Blo);
  aOff += 64; bOff += 64;

  unsigned cur = 0;
  for (int k0 = 0; k0 < KDIM; k0 += 32) {
    if (k0 + 32 < KDIM) {
      // stage next tile into the other buffer, then wait only for current
      const unsigned nb = (cur ^ 1u) * BUF_BYTES;
      stage_async(ldsAh + nb, ldsAl + nb, ldsBh + nb, ldsBl + nb,
                  aOff, bOff, Ahi, Alo, Bhi, Blo);
      aOff += 64; bOff += 64;
      asm volatile("s_wait_asynccnt 0x8" ::: "memory");  // current tile landed
    } else {
      asm volatile("s_wait_asynccnt 0x0" ::: "memory");
    }
    __syncthreads();

    const unsigned short* buf = smem + (size_t)cur * BUF_HALVES;
    const unsigned short (*sAh)[LDS_ROW] = (const unsigned short(*)[LDS_ROW])(buf);
    const unsigned short (*sAl)[LDS_ROW] = (const unsigned short(*)[LDS_ROW])(buf + SMEM_HALVES);
    const unsigned short (*sBh)[LDS_ROW] = (const unsigned short(*)[LDS_ROW])(buf + 2 * SMEM_HALVES);
    const unsigned short (*sBl)[LDS_ROW] = (const unsigned short(*)[LDS_ROW])(buf + 3 * SMEM_HALVES);

    // A fragment (16-bit 16x32, ISA 7.12.2): lane holds row fr;
    // halves[0..7] = K[8*hk .. 8*hk+7], halves[8..15] = K[8*hk+16 .. +23]
    BFV aH[2], aL[2];
#pragma unroll
    for (int mi = 0; mi < 2; ++mi) {
      const int r  = wm * 32 + mi * 16 + fr;
      const int ka = hk * 8;
      aH[mi].h[0] = *(const v8bf*)&sAh[r][ka];
      aH[mi].h[1] = *(const v8bf*)&sAh[r][ka + 16];
      aL[mi].h[0] = *(const v8bf*)&sAl[r][ka];
      aL[mi].h[1] = *(const v8bf*)&sAl[r][ka + 16];
    }
    // B fragment (32x16): lane holds column fr; halves[0..7] = K[8*hk .. 8*hk+7], halves[8..15] = K[8*hk+16 .. +23]
    BFV bH[4], bL[4];
#pragma unroll
    for (int ni = 0; ni < 4; ++ni) {
      const int r  = wn * 64 + ni * 16 + fr;
      const int kb = hk * 8;
      bH[ni].h[0] = *(const v8bf*)&sBh[r][kb];
      bH[ni].h[1] = *(const v8bf*)&sBh[r][kb + 16];
      bL[ni].h[0] = *(const v8bf*)&sBl[r][kb];
      bL[ni].h[1] = *(const v8bf*)&sBl[r][kb + 16];
    }

#pragma unroll
    for (int mi = 0; mi < 2; ++mi)
#pragma unroll
      for (int ni = 0; ni < 4; ++ni) {
        acc[mi][ni] = __builtin_amdgcn_wmma_f32_16x16x32_bf16(
            false, aH[mi].v, false, bH[ni].v, (short)0, acc[mi][ni], false, false);
        acc[mi][ni] = __builtin_amdgcn_wmma_f32_16x16x32_bf16(
            false, aH[mi].v, false, bL[ni].v, (short)0, acc[mi][ni], false, false);
        acc[mi][ni] = __builtin_amdgcn_wmma_f32_16x16x32_bf16(
            false, aL[mi].v, false, bH[ni].v, (short)0, acc[mi][ni], false, false);
      }
    __syncthreads();

    cur ^= 1u;
  }

  // D layout (32-bit 16x16): element (m = r + 8*hk, n = fr) in VGPR r
#pragma unroll
  for (int mi = 0; mi < 2; ++mi)
#pragma unroll
    for (int ni = 0; ni < 4; ++ni) {
      const int col = n0 + wn * 64 + ni * 16 + fr;
#pragma unroll
      for (int r = 0; r < 8; ++r) {
        const int row = m0 + wm * 32 + mi * 16 + r + hk * 8;
        C[(size_t)row * NTOT + col] = acc[mi][ni][r];
      }
    }
}

// ---- epilogue: one block per batch row -------------------------------------
__global__ __launch_bounds__(256) void onlstm_epilogue(
    const float* __restrict__ gates,     // (BSZ, NTOT)
    const float* __restrict__ c_prev,    // (BSZ, HID)
    float* __restrict__ out) {           // h_new (BSZ,HID) then c_new (BSZ,HID)
  const int b = blockIdx.x;
  const int t = threadIdx.x;
  const float* grow = gates + (size_t)b * NTOT;

  __shared__ float sq[LVL], sp[LVL], iH[LVL], fH[LVL], red[LVL];

  float eq = 0.f;
  if (t < LVL) {
    sq[t] = grow[NLSTM + t];          // cc_i_h
    sp[t] = grow[NLSTM + LVL + t];    // cc_f_h
  }
  __syncthreads();

  // max(cc_i_h)
  if (t < LVL) red[t] = sq[t];
  __syncthreads();
  for (int s = LVL / 2; s > 0; s >>= 1) {
    if (t < s) red[t] = fmaxf(red[t], red[t + s]);
    __syncthreads();
  }
  const float mq = red[0];
  __syncthreads();
  // max(cc_f_h)
  if (t < LVL) red[t] = sp[t];
  __syncthreads();
  for (int s = LVL / 2; s > 0; s >>= 1) {
    if (t < s) red[t] = fmaxf(red[t], red[t + s]);
    __syncthreads();
  }
  const float mp = red[0];
  __syncthreads();

  if (t < LVL) {
    eq = __expf(sq[t] - mq);
    sq[t] = eq;
    sp[t] = __expf(sp[t] - mp);
  }
  __syncthreads();

  // inclusive scans (Hillis-Steele) of both exp arrays
  for (int s = 1; s < LVL; s <<= 1) {
    float vq = 0.f, vp = 0.f;
    if (t < LVL && t >= s) { vq = sq[t - s]; vp = sp[t - s]; }
    __syncthreads();
    if (t < LVL) { sq[t] += vq; sp[t] += vp; }
    __syncthreads();
  }
  const float tq = sq[LVL - 1];
  const float tp = sp[LVL - 1];
  if (t < LVL) {
    iH[t] = sp[t] / tp;                 // cumsum(softmax(cc_f_h))
    fH[t] = (tq - sq[t] + eq) / tq;     // reversed-cumsum identity for cc_i_h
  }
  __syncthreads();

  for (int j = t; j < HID; j += 256) {
    const float cci = grow[j];
    const float ccf = grow[HID + j];
    const float cco = grow[2 * HID + j];
    const float ccg = grow[3 * HID + j];
    const float iv = 1.f / (1.f + __expf(-cci));
    const float fv = 1.f / (1.f + __expf(-ccf));
    const float ov = 1.f / (1.f + __expf(-cco));
    const float gv = tanhf(ccg);
    const float cp = c_prev[(size_t)b * HID + j];
    const int   l  = j >> 4;            // N_REPEAT = 16
    const float ih = iH[l];
    const float fh = fH[l];
    const float w  = ih * fh;
    const float cn = w * (fv * cp + iv * gv) + (fh - w) * cp + (ih - w) * gv;
    const float hn = ov * tanhf(cn);
    out[(size_t)b * HID + j] = hn;
    out[(size_t)BSZ * HID + (size_t)b * HID + j] = cn;
  }
}

// ---------------------------------------------------------------------------
extern "C" void kernel_launch(void* const* d_in, const int* in_sizes, int n_in,
                              void* d_out, int out_size, void* d_ws, size_t ws_size,
                              hipStream_t stream) {
  (void)in_sizes; (void)n_in; (void)out_size; (void)ws_size;
  const float* input   = (const float*)d_in[0];
  const float* h_prev  = (const float*)d_in[1];
  const float* c_prev  = (const float*)d_in[2];
  const float* W_lstm  = (const float*)d_in[3];
  const float* W_level = (const float*)d_in[4];

  unsigned short* Ahi = (unsigned short*)d_ws;
  unsigned short* Alo = Ahi + (size_t)BSZ * KDIM;
  unsigned short* Bhi = Alo + (size_t)BSZ * KDIM;
  unsigned short* Blo = Bhi + (size_t)NTOT * KDIM;
  float*          gts = (float*)(Blo + (size_t)NTOT * KDIM);
  float*          out = (float*)d_out;

  {
    const size_t n = (size_t)BSZ * KDIM / 4;
    pack_A<<<dim3((unsigned)((n + 255) / 256)), dim3(256), 0, stream>>>(
        input, h_prev, Ahi, Alo);
  }
  {
    const size_t n = (size_t)NTOT * KDIM / 4;
    pack_B<<<dim3((unsigned)((n + 255) / 256)), dim3(256), 0, stream>>>(
        W_lstm, W_level, Bhi, Blo);
  }
  gemm_bf16x3<<<dim3(NTOT / 128, BSZ / 128), dim3(256), 0, stream>>>(
      Ahi, Alo, Bhi, Blo, gts);
  onlstm_epilogue<<<dim3(BSZ), dim3(256), 0, stream>>>(gts, c_prev, out);
}
